// MemoryLayer_85392539779863
// MI455X (gfx1250) — matrix operand order repaired and checked by reading, not compiled, not run on hardware
//
#include <hip/hip_runtime.h>

// ---------------------------------------------------------------------------
// MemoryLayer fused retrieval for MI455X (gfx1250, wave32, WMMA bf16).
//
//   k_init      : zero max_w accumulator + depth-max cell
//   k_lgd       : lgd[n] = log(depths)+log(gate); atomic-max of depths
//   k_transpose : patterns f32 -> bf16 row-major (pb) + bf16 transposed (pT)
//   k_project   : projected = state @ W^T in f32, rounded once to bf16
//   k_flash     : flash-softmax fused GEMM1+softmax+GEMM2 (v_wmma bf16),
//                 grid (64 batch tiles x 8 N segments), partials to ws
//   k_combine   : cross-segment flash merge -> retrieved, mfin, lfin
//   k_maxw      : recompute sims (WMMA), column-max of weights -> max_w
//   k_depths    : new_depths elementwise
// ---------------------------------------------------------------------------

typedef __attribute__((ext_vector_type(16))) __bf16 v16bf;
typedef __attribute__((ext_vector_type(8)))  float  v8f;

#define BATCH 1024
#define DDIM  256
#define NPAT  100000
#define NW    8            // waves per workgroup in the WMMA kernels
#define SEG   8            // N segments (grid.y)
#define NCHUNK (NPAT / 32) // 3125 chunks of 32 patterns

__device__ __forceinline__ v8f wmma_bf16(v16bf a, v16bf b, v8f c) {
  return __builtin_amdgcn_wmma_f32_16x16x32_bf16(
      false, a, false, b, (short)0, c, false, false);
}

__device__ __forceinline__ v16bf ld16(const __bf16* p) {
  v16bf r;
  ((float4*)&r)[0] = *(const float4*)(p);
  ((float4*)&r)[1] = *(const float4*)(p + 8);
  return r;
}

// ---------------------------------------------------------------- prep ------
__global__ void k_init(unsigned* __restrict__ maxw, unsigned* __restrict__ dmax,
                       int n) {
  int i = blockIdx.x * blockDim.x + threadIdx.x;
  if (i < n) maxw[i] = 0u;
  if (i == 0) *dmax = 0u;
}

__global__ void k_lgd(const float* __restrict__ depths,
                      const float* __restrict__ gate,
                      float* __restrict__ lgd, unsigned* __restrict__ dmax,
                      int n) {
  int i = blockIdx.x * blockDim.x + threadIdx.x;
  if (i >= n) return;
  float d = depths[i];
  lgd[i] = logf(fmaxf(d, 1e-8f)) + logf(fmaxf(gate[i], 1e-8f));
  atomicMax(dmax, __float_as_uint(d));  // depths > 0 -> uint order == fp order
}

__global__ void k_transpose(const float* __restrict__ pat,
                            __bf16* __restrict__ pb,
                            __bf16* __restrict__ pT) {
  __shared__ __bf16 tile[32][33];
  const int n0 = blockIdx.x * 32;
  const int d0 = blockIdx.y * 32;
  const int tx = threadIdx.x, ty = threadIdx.y;  // block (32,8)
  for (int r = ty; r < 32; r += 8) {
    float v = pat[(size_t)(n0 + r) * DDIM + d0 + tx];
    __bf16 b = (__bf16)v;
    pb[(size_t)(n0 + r) * DDIM + d0 + tx] = b;
    tile[r][tx] = b;
  }
  __syncthreads();
  for (int r = ty; r < 32; r += 8)
    pT[(size_t)(d0 + r) * NPAT + n0 + tx] = tile[tx][r];
}

__global__ void k_project(const float* __restrict__ state,
                          const float* __restrict__ W,
                          __bf16* __restrict__ projB) {
  int idx = blockIdx.x * blockDim.x + threadIdx.x;
  if (idx >= BATCH * DDIM) return;
  int b = idx >> 8, d = idx & 255;
  const float* s = state + b * DDIM;
  const float* w = W + d * DDIM;     // y = x @ W^T
  float acc = 0.f;
#pragma unroll 8
  for (int k = 0; k < DDIM; ++k) acc = fmaf(s[k], w[k], acc);
  projB[idx] = (__bf16)acc;
}

// ------------------------------------------------------- flash main pass ----
__global__ __launch_bounds__(256)
__attribute__((amdgpu_waves_per_eu(1)))
void k_flash(const __bf16* __restrict__ projB,  // [B][256] bf16
             const __bf16* __restrict__ pb,     // [N][256] bf16
             const __bf16* __restrict__ pT,     // [256][N] bf16
             const float* __restrict__ lgd,     // [N]
             float* __restrict__ macc,          // [64*SEG][16][256] partials
             float* __restrict__ pm, float* __restrict__ pl) {
  const int wave = threadIdx.x >> 5;
  const int lane = threadIdx.x & 31;
  const int h    = lane >> 4;
  const int l15  = lane & 15;
  const int rowBase = blockIdx.x * 16;                 // batch tile
  const int pidx    = blockIdx.x * SEG + blockIdx.y;   // partial slot

  __shared__ __bf16 ldsP[NW][16][32];
  __shared__ float  ldsM[NW][16];
  __shared__ float  ldsL[NW][16];
  __shared__ float  accsum[16][DDIM];

  // Persistent A fragments (projected tile 16x256, 8 K-steps of 32).
  v16bf afrag[8];
  {
    const __bf16* aptr = projB + (size_t)(rowBase + l15) * DDIM;
#pragma unroll
    for (int dd = 0; dd < 8; ++dd) {
      ((float4*)&afrag[dd])[0] = *(const float4*)(aptr + dd * 32 + 16 * h);
      ((float4*)&afrag[dd])[1] = *(const float4*)(aptr + dd * 32 + 16 * h + 8);
    }
  }

  const v8f vzero = {0.f, 0.f, 0.f, 0.f, 0.f, 0.f, 0.f, 0.f};
  v8f acc[16];
#pragma unroll
  for (int t = 0; t < 16; ++t) acc[t] = vzero;
  float m[8], lsum[8];
#pragma unroll
  for (int j = 0; j < 8; ++j) { m[j] = -1e30f; lsum[j] = 0.f; }

  for (int c = blockIdx.y * NW + wave; c < NCHUNK; c += SEG * NW) {
    const int n0 = c * 32;
    // ---- GEMM1: sims tiles S0/S1, double-buffered B fragments ------------
    v8f S0 = vzero, S1 = vzero;
    const __bf16* b0 = pb + (size_t)(n0 + l15) * DDIM + 16 * h;
    const __bf16* b1 = pb + (size_t)(n0 + 16 + l15) * DDIM + 16 * h;
    v16bf cur0 = ld16(b0), cur1 = ld16(b1);
#pragma unroll
    for (int dd = 0; dd < 8; ++dd) {
      v16bf nxt0, nxt1;
      if (dd < 7) {
        nxt0 = ld16(b0 + (dd + 1) * 32);
        nxt1 = ld16(b1 + (dd + 1) * 32);
      }
      S0 = wmma_bf16(afrag[dd], cur0, S0);
      S1 = wmma_bf16(afrag[dd], cur1, S1);
      cur0 = nxt0; cur1 = nxt1;
    }
    const float g0 = lgd[n0 + l15];
    const float g1 = lgd[n0 + 16 + l15];

    // ---- online softmax per row pair (row = j + 8h) ----------------------
#pragma unroll
    for (int j = 0; j < 8; ++j) {
      float s0 = S0[j] + g0, s1 = S1[j] + g1;
      float mj = fmaxf(s0, s1);
      mj = fmaxf(mj, __shfl_xor(mj, 1));
      mj = fmaxf(mj, __shfl_xor(mj, 2));
      mj = fmaxf(mj, __shfl_xor(mj, 4));
      mj = fmaxf(mj, __shfl_xor(mj, 8));
      float mn = fmaxf(m[j], mj);
      float scale = __expf(m[j] - mn);
      m[j] = mn;
      float e0 = __expf(s0 - mn), e1 = __expf(s1 - mn);
      float rs = e0 + e1;
      rs += __shfl_xor(rs, 1);
      rs += __shfl_xor(rs, 2);
      rs += __shfl_xor(rs, 4);
      rs += __shfl_xor(rs, 8);
      lsum[j] = lsum[j] * scale + rs;
#pragma unroll
      for (int t = 0; t < 16; ++t) acc[t][j] *= scale;
      ldsP[wave][j + 8 * h][l15]      = (__bf16)e0;
      ldsP[wave][j + 8 * h][l15 + 16] = (__bf16)e1;
    }

    // ---- GEMM2: acc += P(16x32) @ patterns_chunk, double-buffered --------
    v16bf pa;   // same-wave DS ops are in-order
    ((float4*)&pa)[0] = *(const float4*)&ldsP[wave][l15][16 * h];
    ((float4*)&pa)[1] = *(const float4*)&ldsP[wave][l15][16 * h + 8];
    const __bf16* bpBase = pT + (size_t)l15 * NPAT + n0 + 16 * h;
    v16bf bcur = ld16(bpBase);
#pragma unroll
    for (int t = 0; t < 16; ++t) {
      v16bf bnxt;
      if (t < 15) bnxt = ld16(bpBase + (size_t)(t + 1) * 16 * NPAT);
      acc[t] = wmma_bf16(pa, bcur, acc[t]);
      bcur = bnxt;
    }
  }

  // ---- cross-wave merge within the WG ------------------------------------
  if (l15 == 0) {
#pragma unroll
    for (int j = 0; j < 8; ++j) ldsM[wave][j + 8 * h] = m[j];
  }
  __syncthreads();
  float scl[8];
#pragma unroll
  for (int j = 0; j < 8; ++j) {
    float mm = -1e30f;
    for (int w = 0; w < NW; ++w) mm = fmaxf(mm, ldsM[w][j + 8 * h]);
    scl[j] = __expf(m[j] - mm);
    if (l15 == 0) ldsL[wave][j + 8 * h] = lsum[j] * scl[j];
  }
  for (int i = threadIdx.x; i < 16 * DDIM; i += blockDim.x)
    ((float*)accsum)[i] = 0.f;
  __syncthreads();
#pragma unroll
  for (int j = 0; j < 8; ++j) {
    float s = scl[j];
#pragma unroll
    for (int t = 0; t < 16; ++t)
      atomicAdd(&accsum[j + 8 * h][t * 16 + l15], acc[t][j] * s);
  }
  __syncthreads();
  if (threadIdx.x < 16) {
    float L = 0.f, mm = -1e30f;
    for (int w = 0; w < NW; ++w) {
      L += ldsL[w][threadIdx.x];
      mm = fmaxf(mm, ldsM[w][threadIdx.x]);
    }
    pm[pidx * 16 + threadIdx.x] = mm;
    pl[pidx * 16 + threadIdx.x] = L;
  }
  __syncthreads();
  for (int i = threadIdx.x; i < 16 * DDIM; i += blockDim.x) {
    int r = i >> 8, d = i & 255;
    macc[((size_t)pidx * 16 + r) * DDIM + d] = accsum[r][d];
  }
}

// ------------------------------------------- cross-segment flash merge -----
__global__ void k_combine(const float* __restrict__ macc,
                          const float* __restrict__ pm,
                          const float* __restrict__ pl,
                          float* __restrict__ out, float* __restrict__ mfin,
                          float* __restrict__ lfin) {
  __shared__ float sM[16], sL[16], sc[SEG][16];
  const int btile = blockIdx.x;
  const int tid = threadIdx.x;
  if (tid < 16) {
    float mm = -1e30f;
    for (int s = 0; s < SEG; ++s)
      mm = fmaxf(mm, pm[(btile * SEG + s) * 16 + tid]);
    sM[tid] = mm;
  }
  __syncthreads();
  if (tid < 16 * SEG) {
    int s = tid >> 4, r = tid & 15;
    sc[s][r] = __expf(pm[(btile * SEG + s) * 16 + r] - sM[r]);
  }
  __syncthreads();
  if (tid < 16) {
    float L = 0.f;
    for (int s = 0; s < SEG; ++s)
      L += pl[(btile * SEG + s) * 16 + tid] * sc[s][tid];
    sL[tid] = L;
    mfin[btile * 16 + tid] = sM[tid];
    lfin[btile * 16 + tid] = L;
  }
  __syncthreads();
  for (int i = tid; i < 16 * DDIM; i += blockDim.x) {
    int r = i >> 8, d = i & 255;
    float v = 0.f;
    for (int s = 0; s < SEG; ++s)
      v += macc[((size_t)(btile * SEG + s) * 16 + r) * DDIM + d] * sc[s][r];
    out[(size_t)(btile * 16 + r) * DDIM + d] = v / sL[r];
  }
}

// ------------------------------------------------ max_w recompute pass -----
__global__ __launch_bounds__(256)
__attribute__((amdgpu_waves_per_eu(1)))
void k_maxw(const __bf16* __restrict__ projB, const __bf16* __restrict__ pb,
            const float* __restrict__ lgd, const float* __restrict__ mfin,
            const float* __restrict__ lfin, unsigned* __restrict__ maxw) {
  const int wave = threadIdx.x >> 5;
  const int lane = threadIdx.x & 31;
  const int h    = lane >> 4;
  const int l15  = lane & 15;
  const int rowBase = blockIdx.x * 16;

  v16bf afrag[8];
  {
    const __bf16* aptr = projB + (size_t)(rowBase + l15) * DDIM;
#pragma unroll
    for (int dd = 0; dd < 8; ++dd) {
      ((float4*)&afrag[dd])[0] = *(const float4*)(aptr + dd * 32 + 16 * h);
      ((float4*)&afrag[dd])[1] = *(const float4*)(aptr + dd * 32 + 16 * h + 8);
    }
  }
  float mrow[8], inv[8];
#pragma unroll
  for (int j = 0; j < 8; ++j) {
    mrow[j] = mfin[rowBase + j + 8 * h];
    inv[j]  = 1.f / lfin[rowBase + j + 8 * h];
  }
  const v8f vzero = {0.f, 0.f, 0.f, 0.f, 0.f, 0.f, 0.f, 0.f};

  for (int c = blockIdx.y * NW + wave; c < NCHUNK; c += SEG * NW) {
    const int n0 = c * 32;
    v8f S0 = vzero, S1 = vzero;
    const __bf16* b0 = pb + (size_t)(n0 + l15) * DDIM + 16 * h;
    const __bf16* b1 = pb + (size_t)(n0 + 16 + l15) * DDIM + 16 * h;
    v16bf cur0 = ld16(b0), cur1 = ld16(b1);
#pragma unroll
    for (int dd = 0; dd < 8; ++dd) {
      v16bf nxt0, nxt1;
      if (dd < 7) {
        nxt0 = ld16(b0 + (dd + 1) * 32);
        nxt1 = ld16(b1 + (dd + 1) * 32);
      }
      S0 = wmma_bf16(afrag[dd], cur0, S0);
      S1 = wmma_bf16(afrag[dd], cur1, S1);
      cur0 = nxt0; cur1 = nxt1;
    }
    const float g0 = lgd[n0 + l15];
    const float g1 = lgd[n0 + 16 + l15];
    float cm0 = 0.f, cm1 = 0.f;
#pragma unroll
    for (int j = 0; j < 8; ++j) {
      cm0 = fmaxf(cm0, __expf(S0[j] + g0 - mrow[j]) * inv[j]);
      cm1 = fmaxf(cm1, __expf(S1[j] + g1 - mrow[j]) * inv[j]);
    }
    cm0 = fmaxf(cm0, __shfl_xor(cm0, 16));
    cm1 = fmaxf(cm1, __shfl_xor(cm1, 16));
    if (h == 0) {  // weights >= 0 -> uint atomic-max is exact & deterministic
      atomicMax(&maxw[n0 + l15],      __float_as_uint(cm0));
      atomicMax(&maxw[n0 + 16 + l15], __float_as_uint(cm1));
    }
  }
}

// ----------------------------------------------------- depth update --------
__global__ void k_depths(const float* __restrict__ depths,
                         const unsigned* __restrict__ maxw,
                         const unsigned* __restrict__ dmax,
                         float* __restrict__ out, int n) {
  int i = blockIdx.x * blockDim.x + threadIdx.x;
  if (i >= n) return;
  float d  = depths[i];
  float mw = __uint_as_float(maxw[i]);
  float ldm = log1pf(__uint_as_float(*dmax));
  float dom = log1pf(d) / fmaxf(ldm, 1e-8f);
  float rate = 0.01f * (1.f - 0.7f * dom);
  out[i] = d + ((mw > 0.1f) ? rate * mw : 0.f);
}

// ---------------------------------------------------------------------------
extern "C" void kernel_launch(void* const* d_in, const int* in_sizes, int n_in,
                              void* d_out, int out_size, void* d_ws,
                              size_t ws_size, hipStream_t stream) {
  const float* state    = (const float*)d_in[0];
  const float* W        = (const float*)d_in[1];
  const float* patterns = (const float*)d_in[2];
  const float* depths   = (const float*)d_in[3];
  const float* gate     = (const float*)d_in[4];
  float* out = (float*)d_out;

  char* ws = (char*)d_ws;
  const size_t PB  = (size_t)NPAT * DDIM * 2;      // 51.2 MB bf16 patterns
  size_t off = 0;
  __bf16* pb    = (__bf16*)(ws + off); off += PB;
  __bf16* pT    = (__bf16*)(ws + off); off += PB;
  __bf16* projB = (__bf16*)(ws + off); off += (size_t)BATCH * DDIM * 2;
  float*  lgd   = (float*)(ws + off);  off += 400128;
  unsigned* maxw = (unsigned*)(ws + off); off += 400128;
  float*  mfin  = (float*)(ws + off);  off += 4096;
  float*  lfin  = (float*)(ws + off);  off += 4096;
  unsigned* dmax = (unsigned*)(ws + off); off += 256;
  float*  macc  = (float*)(ws + off);  off += (size_t)64 * SEG * 16 * DDIM * 4;
  float*  pm    = (float*)(ws + off);  off += (size_t)64 * SEG * 16 * 4;
  float*  pl    = (float*)(ws + off);  off += (size_t)64 * SEG * 16 * 4;

  const int nb = (NPAT + 255) / 256;
  k_init<<<nb, 256, 0, stream>>>(maxw, dmax, NPAT);
  k_lgd<<<nb, 256, 0, stream>>>(depths, gate, lgd, dmax, NPAT);
  k_transpose<<<dim3(NPAT / 32, DDIM / 32), dim3(32, 8), 0, stream>>>(
      patterns, pb, pT);
  k_project<<<(BATCH * DDIM) / 256, 256, 0, stream>>>(state, W, projB);
  k_flash<<<dim3(BATCH / 16, SEG), 256, 0, stream>>>(projB, pb, pT, lgd, macc,
                                                     pm, pl);
  k_combine<<<BATCH / 16, 256, 0, stream>>>(macc, pm, pl, out, mfin, lfin);
  k_maxw<<<dim3(BATCH / 16, SEG), 256, 0, stream>>>(projB, pb, lgd, mfin, lfin,
                                                    maxw);
  k_depths<<<nb, 256, 0, stream>>>(depths, maxw, dmax, out + BATCH * DDIM,
                                   NPAT);
}
